// linear_attention_pool_1322849927772
// MI455X (gfx1250) — matrix operand order repaired and checked by reading, not compiled, not run on hardware
//
#include <hip/hip_runtime.h>

// ---------------- problem constants ----------------
#define BATCH   16
#define DCH     512          // D (channels)
#define TSEQ    4096         // T
#define NSEQ    4608         // N = D + T
#define MROWS   1024         // rows of Wkv (k then v)
#define HEADS   8
#define DH      64           // dim_head
#define SCALE   0.125f       // 1/sqrt(64)

// WMMA vector types (probe-verified signatures)
typedef __attribute__((ext_vector_type(16))) _Float16 v16h;
typedef __attribute__((ext_vector_type(8)))  float    v8f;

union HFrag { v16h v; float4 f[2]; };

// ---------------- workspace layout ----------------
// [0,       32KB)   : m      (B*512 f32)  mean of q over T
// [32KB,    64KB)   : denom  (B*512 f32)  sum_n exp(k)
// [64KB,    2MB+64K): ctx    (B*8*64*64 f32)
// [2MB+64K, ...)    : kv     (B*1024*4608 f16)
#define WS_M_OFF     0
#define WS_DEN_OFF   32768
#define WS_CTX_OFF   65536
#define WS_KV_OFF    (65536 + 2097152)
#define CTX_BYTES    (BATCH * HEADS * DH * DH * 4)

// =====================================================================
// 1) mean over T:  m[b,d] = mean_t q[b,d,t]
// =====================================================================
__global__ __launch_bounds__(256)
void mean_kernel(const float* __restrict__ q, float* __restrict__ m) {
    __shared__ float red[256];
    const int row = blockIdx.x;              // b*512 + d
    const int t   = threadIdx.x;
    const float* src = q + (size_t)row * TSEQ;
    float s = 0.f;
    for (int i = t; i < TSEQ; i += 256) s += src[i];
    red[t] = s;
    __syncthreads();
    for (int off = 128; off > 0; off >>= 1) {
        if (t < off) red[t] += red[t + off];
        __syncthreads();
    }
    if (t == 0) m[row] = red[0] * (1.f / (float)TSEQ);
}

// =====================================================================
// 2) kv GEMM:  kv[b] (1024 x 4608) = Wkv (1024x512) @ x_b (512x4608)
//    x_b[:, n<512]  = m[b,:] + pe[n]   (synthesized, never materialized)
//    x_b[:, n>=512] = q[b,:, n-512]
//    f16 WMMA, f32 accumulate, store f16.
//    Workgroup tile 128M x 128N, BK=32; 8 waves in 2(M) x 4(N).
// =====================================================================
#define LDP 40   // LDS row pitch (f16 elems): conflict-free frag fetches
__global__ __launch_bounds__(256)
void kv_gemm_kernel(const float* __restrict__ q,
                    const float* __restrict__ pe,
                    const float* __restrict__ Wkv,
                    const float* __restrict__ mvec,
                    _Float16* __restrict__ kv) {
    __shared__ _Float16 ldsA[128 * LDP];   // A tile: [m][k]
    __shared__ _Float16 ldsB[128 * LDP];   // B tile transposed: [n][k]

    const int mBase = blockIdx.x * 128;
    const int nBase = blockIdx.y * 128;
    const int b     = blockIdx.z;
    const int t     = threadIdx.x;
    const int wid   = t >> 5;
    const int lane  = t & 31;
    const int half  = lane >> 4;           // 0 | 1
    const int mr    = lane & 15;
    const int waveM = wid >> 2;            // 0..1  (64 rows each)
    const int waveN = wid & 3;             // 0..3  (32 cols each)

    v8f acc[4][2];
    for (int i = 0; i < 4; ++i)
        for (int j = 0; j < 2; ++j) acc[i][j] = (v8f){};

    for (int kt = 0; kt < DCH / 32; ++kt) {
        // ---- load A tile (Wkv 128x32 f32 -> f16) ----
        {
            const int c4 = t & 7;          // col group of 4
            const int r0 = t >> 3;         // 0..31
            for (int p = 0; p < 4; ++p) {
                const int row = p * 32 + r0;
                const float4 w4 = *reinterpret_cast<const float4*>(
                    &Wkv[(size_t)(mBase + row) * DCH + kt * 32 + c4 * 4]);
                _Float16* dst = &ldsA[row * LDP + c4 * 4];
                dst[0] = (_Float16)w4.x; dst[1] = (_Float16)w4.y;
                dst[2] = (_Float16)w4.z; dst[3] = (_Float16)w4.w;
            }
        }
        // ---- load B tile transposed into [n][k] ----
        for (int p = 0; p < 16; ++p) {
            const int e  = p * 256 + t;
            const int kk = e >> 7;         // 0..31
            const int nn = e & 127;
            const int ng = nBase + nn;
            const int kg = kt * 32 + kk;
            float v;
            if (ng < DCH)
                v = mvec[b * DCH + kg] + pe[ng];           // class-token block
            else
                v = q[((size_t)(b * DCH + kg)) * TSEQ + (ng - DCH)];
            ldsB[nn * LDP + kk] = (_Float16)v;
        }
        __syncthreads();

        // ---- WMMA: 4(M) x 2(N) tiles per wave ----
        HFrag fa[4], fb[2];
        for (int tm = 0; tm < 4; ++tm) {
            const _Float16* ap =
                &ldsA[(waveM * 64 + tm * 16 + mr) * LDP + half * 8];
            fa[tm].f[0] = *reinterpret_cast<const float4*>(ap);
            fa[tm].f[1] = *reinterpret_cast<const float4*>(ap + 16);
        }
        for (int tn = 0; tn < 2; ++tn) {
            const _Float16* bp =
                &ldsB[(waveN * 32 + tn * 16 + mr) * LDP + half * 8];
            fb[tn].f[0] = *reinterpret_cast<const float4*>(bp);
            fb[tn].f[1] = *reinterpret_cast<const float4*>(bp + 16);
        }
        for (int tm = 0; tm < 4; ++tm)
            for (int tn = 0; tn < 2; ++tn)
                acc[tm][tn] = __builtin_amdgcn_wmma_f32_16x16x32_f16(
                    false, fa[tm].v, false, fb[tn].v,
                    (short)0, acc[tm][tn], false, false);
        __syncthreads();
    }

    // ---- store C as f16 ----
    for (int tm = 0; tm < 4; ++tm)
        for (int tn = 0; tn < 2; ++tn)
            for (int r = 0; r < 8; ++r) {
                const int rowG = mBase + waveM * 64 + tm * 16 + half * 8 + r;
                const int colG = nBase + waveN * 32 + tn * 16 + mr;
                kv[((size_t)b * MROWS + rowG) * NSEQ + colG] =
                    (_Float16)acc[tm][tn][r];
            }
}

// =====================================================================
// 3) denom[b,r] = sum_n exp(k[b,r,n])   (k = kv rows 0..511)
//    No max-subtraction: |k| is O(5), exp is safe in f32.
// =====================================================================
__global__ __launch_bounds__(256)
void denom_kernel(const _Float16* __restrict__ kv, float* __restrict__ denom) {
    __shared__ float red[256];
    const int row = blockIdx.x;            // b*512 + r
    const int b   = row >> 9;
    const int r   = row & 511;
    const int t   = threadIdx.x;
    const _Float16* src = kv + ((size_t)b * MROWS + r) * NSEQ;
    float s = 0.f;
    for (int i = t; i < NSEQ; i += 256) s += __expf((float)src[i]);
    red[t] = s;
    __syncthreads();
    for (int off = 128; off > 0; off >>= 1) {
        if (t < off) red[t] += red[t + off];
        __syncthreads();
    }
    if (t == 0) denom[row] = red[0];
}

// =====================================================================
// 4) context_raw[b,h,d,e] += sum_{n in chunk} exp(k[h*64+d,n]) * v[h*64+e,n]
//    64x64 per (b,h), WMMA with exp applied to A on the fly;
//    grid.x splits n into 8 chunks of 576, f32 atomic accumulation.
//    128 threads = 4 waves; wave w owns d-strip [16w,16w+16) x all 64 e.
// =====================================================================
__global__ __launch_bounds__(128)
void context_kernel(const _Float16* __restrict__ kv, float* __restrict__ ctx) {
    const int chunk = blockIdx.x;          // 0..7
    const int h     = blockIdx.y;
    const int b     = blockIdx.z;
    const int t     = threadIdx.x;
    const int wid   = t >> 5;              // 0..3
    const int lane  = t & 31;
    const int half  = lane >> 4;
    const int mr    = lane & 15;

    v8f acc[4];
    for (int i = 0; i < 4; ++i) acc[i] = (v8f){};

    const _Float16* kb =
        kv + ((size_t)b * MROWS + h * DH + wid * 16 + mr) * NSEQ;

    for (int s = 0; s < 18; ++s) {
        const int n0 = chunk * 576 + s * 32;
        // A frag: exp(k) for 16 rows of d
        HFrag fa;
        fa.f[0] = *reinterpret_cast<const float4*>(kb + n0 + half * 8);
        fa.f[1] = *reinterpret_cast<const float4*>(kb + n0 + half * 8 + 16);
        for (int i = 0; i < 16; ++i)
            fa.v[i] = (_Float16)__expf((float)fa.v[i]);
        // B frags: v rows (kv rows 512 + h*64 + e), f16 direct
        for (int tn = 0; tn < 4; ++tn) {
            const _Float16* vp =
                kv + ((size_t)b * MROWS + DCH + h * DH + tn * 16 + mr) * NSEQ +
                n0 + half * 8;
            HFrag fb;
            fb.f[0] = *reinterpret_cast<const float4*>(vp);
            fb.f[1] = *reinterpret_cast<const float4*>(vp + 16);
            acc[tn] = __builtin_amdgcn_wmma_f32_16x16x32_f16(
                false, fa.v, false, fb.v, (short)0, acc[tn], false, false);
        }
    }

    for (int tn = 0; tn < 4; ++tn)
        for (int r = 0; r < 8; ++r) {
            const int d = wid * 16 + half * 8 + r;
            const int e = tn * 16 + mr;
            atomicAdd(&ctx[(((size_t)b * HEADS + h) * DH + d) * DH + e],
                      acc[tn][r]);
        }
}

// =====================================================================
// 5) finisher, one block per batch (all tiny f32 matvecs):
//    x0 = m[b] + pe[0];  qh0 = Wq @ x0;  per-head softmax * SCALE;
//    fold 1/denom into weights;  out0 = ctx^T @ w;  y = Wout @ out0 + bout
// =====================================================================
__global__ __launch_bounds__(256)
void final_kernel(const float* __restrict__ pe,
                  const float* __restrict__ Wq,
                  const float* __restrict__ Wout,
                  const float* __restrict__ bout,
                  const float* __restrict__ mvec,
                  const float* __restrict__ denom,
                  const float* __restrict__ ctx,
                  float* __restrict__ out) {
    __shared__ float sx[512], sq[512], sw[512], so[512];
    const int b = blockIdx.x;
    const int t = threadIdx.x;
    const float pe0 = pe[0];

    for (int i = t; i < 512; i += 256) sx[i] = mvec[b * 512 + i] + pe0;
    __syncthreads();

    for (int o = t; o < 512; o += 256) {
        float s = 0.f;
        const float* wr = Wq + (size_t)o * 512;
        for (int d = 0; d < 512; ++d) s += wr[d] * sx[d];
        sq[o] = s;
    }
    __syncthreads();

    if (t < HEADS) {
        const int base = t * DH;
        float mx = -1e30f;
        for (int j = 0; j < DH; ++j) mx = fmaxf(mx, sq[base + j]);
        float sum = 0.f;
        for (int j = 0; j < DH; ++j) sum += __expf(sq[base + j] - mx);
        const float inv = SCALE / sum;
        for (int j = 0; j < DH; ++j)
            sw[base + j] = __expf(sq[base + j] - mx) * inv /
                           denom[b * 512 + base + j];
    }
    __syncthreads();

    for (int he = t; he < 512; he += 256) {
        const int h = he >> 6, e = he & 63;
        const float* cb = ctx + (((size_t)b * HEADS + h) * DH) * DH + e;
        float s = 0.f;
        for (int d = 0; d < DH; ++d) s += cb[d * DH] * sw[h * DH + d];
        so[he] = s;
    }
    __syncthreads();

    for (int o = t; o < 512; o += 256) {
        float s = bout[o];
        const float* wr = Wout + (size_t)o * 512;
        for (int he = 0; he < 512; ++he) s += wr[he] * so[he];
        out[b * 512 + o] = s;
    }
}

// =====================================================================
extern "C" void kernel_launch(void* const* d_in, const int* in_sizes, int n_in,
                              void* d_out, int out_size, void* d_ws, size_t ws_size,
                              hipStream_t stream) {
    const float* q    = (const float*)d_in[0];
    const float* pe   = (const float*)d_in[1];
    const float* Wq   = (const float*)d_in[2];
    const float* Wkv  = (const float*)d_in[3];
    const float* Wout = (const float*)d_in[4];
    const float* bout = (const float*)d_in[5];
    float* out = (float*)d_out;

    char* ws = (char*)d_ws;
    float*    mvec  = (float*)(ws + WS_M_OFF);
    float*    denom = (float*)(ws + WS_DEN_OFF);
    float*    ctx   = (float*)(ws + WS_CTX_OFF);
    _Float16* kv    = (_Float16*)(ws + WS_KV_OFF);

    hipMemsetAsync(ctx, 0, CTX_BYTES, stream);

    mean_kernel<<<BATCH * DCH, 256, 0, stream>>>(q, mvec);

    kv_gemm_kernel<<<dim3(MROWS / 128, NSEQ / 128, BATCH), 256, 0, stream>>>(
        q, pe, Wkv, mvec, kv);

    denom_kernel<<<BATCH * DCH, 256, 0, stream>>>(kv, denom);

    context_kernel<<<dim3(8, HEADS, BATCH), 128, 0, stream>>>(kv, ctx);

    final_kernel<<<BATCH, 256, 0, stream>>>(pe, Wq, Wout, bout,
                                            mvec, denom, ctx, out);
}
